// LSTMCell_scratch_47596827574335
// MI455X (gfx1250) — compile-verified, matrix operand order checked by reading, not run on hardware
//
#include <hip/hip_runtime.h>

// LSTM cell on MI455X (gfx1250), three-phase:
//  1) convert [x|h_prev] f32 -> bf16 A   [4096][2048]   (ws)
//  2) convert+transpose {Wf,Wi,Wg,Wo} -> bf16 W^T [4096][2048] (ws)
//  3) fused GEMM (v_wmma_f32_16x16x32_bf16, f32 accum) + LSTM epilogue,
//     staging via GLOBAL_LOAD_ASYNC_TO_LDS_B128 (ASYNCcnt DMA path).

#define B_DIM  4096
#define IN_DIM 1024
#define H_DIM  1024
#define K_DIM  (IN_DIM + H_DIM)
#define GATES  4

#define BM 128          // rows per workgroup
#define BN 32           // cols per workgroup (per gate)
#define KT 32           // K per WMMA step
#define LDS_ROW 80      // bytes per LDS row: 64B data + 16B pad (conflict-free, 16B aligned)

typedef __attribute__((ext_vector_type(16))) __bf16 v16bf;
typedef __attribute__((ext_vector_type(8)))  __bf16 v8bf;
typedef __attribute__((ext_vector_type(8)))  float  v8f;
typedef __attribute__((ext_vector_type(4)))  int    v4i;

#if __has_builtin(__builtin_amdgcn_global_load_async_to_lds_b128)
#define HAVE_ASYNC_LDS 1
typedef __attribute__((address_space(1))) v4i* gbl_v4i;
typedef __attribute__((address_space(3))) v4i* lds_v4i;
#endif

// ---------------- helpers ----------------

__device__ __forceinline__ unsigned short bf16_rne(float f) {
    unsigned u = __float_as_uint(f);
    u += 0x7FFFu + ((u >> 16) & 1u);
    return (unsigned short)(u >> 16);
}
__device__ __forceinline__ unsigned pack2_bf16(float a, float b) {
    return (unsigned)bf16_rne(a) | ((unsigned)bf16_rne(b) << 16);
}
__device__ __forceinline__ float sigm_f(float x) {
    return 1.0f / (1.0f + __expf(-x));
}
__device__ __forceinline__ float tanh_f(float x) {
    float e = __expf(-2.0f * x);
    return (1.0f - e) / (1.0f + e);
}

// 16-byte global -> LDS stage (async DMA when the toolchain exposes it)
__device__ __forceinline__ void stage_b128(const void* gsrc, void* ldst) {
#ifdef HAVE_ASYNC_LDS
    __builtin_amdgcn_global_load_async_to_lds_b128((gbl_v4i)gsrc, (lds_v4i)ldst,
                                                   /*offset=*/0, /*cpol=*/0);
#else
    *(uint4*)ldst = *(const uint4*)gsrc;
#endif
}

__device__ __forceinline__ void stage_wait() {
#ifdef HAVE_ASYNC_LDS
#if __has_builtin(__builtin_amdgcn_s_wait_asynccnt)
    __builtin_amdgcn_s_wait_asynccnt(0);
#else
    asm volatile("s_wait_asynccnt 0x0" ::: "memory");
#endif
#endif
}

// ---------------- phase 1: A = [x | h_prev] -> bf16, row-major ----------------

__global__ __launch_bounds__(256)
void convert_A_kernel(const float* __restrict__ x,
                      const float* __restrict__ h_prev,
                      unsigned* __restrict__ Abf /* [B][K/2] packed pairs */)
{
    const int tid = threadIdx.x;
    #pragma unroll
    for (int i = 0; i < 4; ++i) {
        const int pair = blockIdx.x * 1024 + i * 256 + tid;  // 0 .. B*K/2-1
        const int m  = pair >> 10;                           // K/2 = 1024
        const int kk = (pair & 1023) * 2;
        float a, b;
        if (kk < IN_DIM) {
            a = x[m * IN_DIM + kk];
            b = x[m * IN_DIM + kk + 1];
        } else {
            a = h_prev[m * H_DIM + kk - IN_DIM];
            b = h_prev[m * H_DIM + kk - IN_DIM + 1];
        }
        Abf[pair] = pack2_bf16(a, b);
    }
}

// ------- phase 2: W^T[col][k] bf16, col = g*1024 + n (LDS-tiled transpose) -------

__global__ __launch_bounds__(256)
void convert_WT_kernel(const float* __restrict__ Wf, const float* __restrict__ Wi,
                       const float* __restrict__ Wg, const float* __restrict__ Wo,
                       unsigned* __restrict__ WT /* [4H][K/2] packed pairs */)
{
    __shared__ unsigned short tile[32][34];  // padded
    const int tid  = threadIdx.x;
    const int col0 = blockIdx.x * 32;        // 0..4095 over 4 gates
    const int k0   = blockIdx.y * 32;        // 0..2047
    const int g    = col0 >> 10;
    const int n0   = col0 & 1023;
    const float* W = (g == 0) ? Wf : (g == 1) ? Wi : (g == 2) ? Wg : Wo;

    const int nc = tid & 31;
    #pragma unroll
    for (int i = 0; i < 4; ++i) {
        const int kr = (tid >> 5) + 8 * i;
        tile[nc][kr] = bf16_rne(W[(k0 + kr) * H_DIM + n0 + nc]);  // coalesced read
    }
    __syncthreads();
    #pragma unroll
    for (int i = 0; i < 2; ++i) {
        const int idx = tid + 256 * i;       // 0..511
        const int nr  = idx >> 4;
        const int kc  = idx & 15;
        const unsigned lo = tile[nr][kc * 2];
        const unsigned hi = tile[nr][kc * 2 + 1];
        WT[(col0 + nr) * (K_DIM / 2) + (k0 >> 1) + kc] = lo | (hi << 16);  // coalesced write
    }
}

// ---------------- phase 3: fused WMMA GEMM + LSTM epilogue ----------------

__global__ __launch_bounds__(256)
void lstm_wmma_bf16_kernel(const unsigned char* __restrict__ Abf, // [B][K] bf16
                           const unsigned char* __restrict__ WT,  // [4H][K] bf16
                           const float* __restrict__ c_prev,
                           const float* __restrict__ bf, const float* __restrict__ bi,
                           const float* __restrict__ bg, const float* __restrict__ bo,
                           float* __restrict__ out)
{
    __shared__ __align__(16) unsigned char ldsA[BM * LDS_ROW];             // 10240 B
    __shared__ __align__(16) unsigned char ldsB[GATES * BN * LDS_ROW];     // 10240 B

    const int tid    = threadIdx.x;
    const int lane   = tid & 31;
    const int wid    = tid >> 5;
    const int wave_m = wid >> 1;   // 0..3 -> 32-row slab
    const int wave_n = wid & 1;    // 0..1 -> 16-col slab
    const int hsel   = lane >> 4;  // lane half (K interleave select)
    const int nl     = lane & 15;

    const int n0 = blockIdx.x * BN;
    const int m0 = blockIdx.y * BM;

    v8f acc[GATES][2];
    #pragma unroll
    for (int g = 0; g < GATES; ++g)
        #pragma unroll
        for (int s = 0; s < 2; ++s)
            #pragma unroll
            for (int e = 0; e < 8; ++e)
                acc[g][s][e] = 0.0f;

    for (int kt = 0; kt < K_DIM / KT; ++kt) {
        const int k0 = kt * KT;

        __syncthreads();  // previous iteration's fragment reads complete

        // ---- stage A tile: 128 rows x 64B, as 512 16-B chunks ----
        #pragma unroll
        for (int i = 0; i < 2; ++i) {
            const int chunk = i * 256 + tid;           // 0..511
            const int row = chunk >> 2, c = chunk & 3;
            stage_b128(Abf + ((size_t)(m0 + row) * K_DIM + k0) * 2 + c * 16,
                       ldsA + row * LDS_ROW + c * 16);
        }
        // ---- stage B tiles (W^T rows): 4 gates x 32 cols x 64B ----
        #pragma unroll
        for (int i = 0; i < 2; ++i) {
            const int chunk = i * 256 + tid;           // 0..511
            const int idx = chunk >> 2, c = chunk & 3; // idx = g*32 + col
            const int gcol = ((idx >> 5) << 10) + n0 + (idx & 31);
            stage_b128(WT + ((size_t)gcol * K_DIM + k0) * 2 + c * 16,
                       ldsB + idx * LDS_ROW + c * 16);
        }

        stage_wait();     // own wave's async copies landed
        __syncthreads();  // everyone's copies landed

        // ---- A fragments (16-bit A 16x32 layout) ----
        v16bf afrag[2];
        #pragma unroll
        for (int ms = 0; ms < 2; ++ms) {
            const int row = wave_m * 32 + ms * 16 + nl;
            const unsigned char* p = ldsA + row * LDS_ROW + hsel * 16;
            v8bf lo = *(const v8bf*)(p);        // K = hsel*8 .. +7
            v8bf hi = *(const v8bf*)(p + 32);   // K = 16 + hsel*8 .. +7
            afrag[ms] = __builtin_shufflevector(lo, hi,
                0, 1, 2, 3, 4, 5, 6, 7, 8, 9, 10, 11, 12, 13, 14, 15);
        }

        // ---- B fragments + 8 WMMAs ----
        #pragma unroll
        for (int g = 0; g < GATES; ++g) {
            const int idx = g * BN + wave_n * 16 + nl;
            const unsigned char* p = ldsB + idx * LDS_ROW + hsel * 16;
            v8bf lo = *(const v8bf*)(p);
            v8bf hi = *(const v8bf*)(p + 32);
            v16bf bfrag = __builtin_shufflevector(lo, hi,
                0, 1, 2, 3, 4, 5, 6, 7, 8, 9, 10, 11, 12, 13, 14, 15);

            acc[g][0] = __builtin_amdgcn_wmma_f32_16x16x32_bf16(
                false, afrag[0], false, bfrag, (short)0, acc[g][0], false, false);
            acc[g][1] = __builtin_amdgcn_wmma_f32_16x16x32_bf16(
                false, afrag[1], false, bfrag, (short)0, acc[g][1], false, false);
        }
    }

    // ---- fused LSTM epilogue ----
    // C layout: VGPR r, lanes 0-15: M=r, N=lane; lanes 16-31: M=8+r, N=lane-16.
    const int n = n0 + wave_n * 16 + nl;
    const float bfv = bf[n];
    const float biv = bi[n];
    const float bgv = bg[n];
    const float bov = bo[n];

    #pragma unroll
    for (int ms = 0; ms < 2; ++ms) {
        const int mbase = m0 + wave_m * 32 + ms * 16 + 8 * hsel;
        #pragma unroll
        for (int r = 0; r < 8; ++r) {
            const int m = mbase + r;
            const float fg = sigm_f(acc[0][ms][r] + bfv);
            const float ig = sigm_f(acc[1][ms][r] + biv);
            const float gg = tanh_f(acc[2][ms][r] + bgv);
            const float og = sigm_f(acc[3][ms][r] + bov);
            const float cp = c_prev[m * H_DIM + n];
            const float cn = fg * cp + ig * gg;
            const float hn = og * tanh_f(cn);
            out[m * H_DIM + n] = hn;                       // h_next
            out[B_DIM * H_DIM + m * H_DIM + n] = cn;       // c_next
        }
    }
}

// ---------------- launcher ----------------

extern "C" void kernel_launch(void* const* d_in, const int* in_sizes, int n_in,
                              void* d_out, int out_size, void* d_ws, size_t ws_size,
                              hipStream_t stream) {
    (void)in_sizes; (void)n_in; (void)out_size; (void)ws_size;

    const float* x      = (const float*)d_in[0];
    const float* h_prev = (const float*)d_in[1];
    const float* c_prev = (const float*)d_in[2];
    const float* Wf     = (const float*)d_in[3];
    const float* bfp    = (const float*)d_in[4];
    const float* Wi     = (const float*)d_in[5];
    const float* bip    = (const float*)d_in[6];
    const float* Wg     = (const float*)d_in[7];
    const float* bgp    = (const float*)d_in[8];
    const float* Wo     = (const float*)d_in[9];
    const float* bop    = (const float*)d_in[10];
    float*       out    = (float*)d_out;

    // workspace: A bf16 (16 MiB) | W^T bf16 (16 MiB)
    unsigned char* ws  = (unsigned char*)d_ws;
    unsigned*      Abf = (unsigned*)ws;
    unsigned*      WT  = (unsigned*)(ws + (size_t)B_DIM * K_DIM * 2);

    convert_A_kernel<<<dim3(B_DIM * K_DIM / 2048), dim3(256), 0, stream>>>(
        x, h_prev, Abf);

    convert_WT_kernel<<<dim3(GATES * H_DIM / 32, K_DIM / 32), dim3(256), 0, stream>>>(
        Wf, Wi, Wg, Wo, WT);

    lstm_wmma_bf16_kernel<<<dim3(H_DIM / BN, B_DIM / BM), dim3(256), 0, stream>>>(
        (const unsigned char*)Abf, (const unsigned char*)WT,
        c_prev, bfp, bip, bgp, bop, out);
}
